// MultiHeadPosAtt_25915832664783
// MI455X (gfx1250) — matrix operand order repaired and checked by reading, not compiled, not run on hardware
//
#include <hip/hip_runtime.h>
#include <hip/hip_fp16.h>
#include <math.h>

// Problem constants (match reference)
#define BB   2
#define NN   4096
#define HH   4
#define HIDD 256
#define VDD  64
#define RANK0 1228   // floor(0.30*(N-1)) ; thresh = 0.5*(sorted[1228]+sorted[1229])

typedef __attribute__((ext_vector_type(16))) _Float16 v16h;
typedef __attribute__((ext_vector_type(8)))  float    v8f;
typedef __attribute__((ext_vector_type(4)))  int      v4i;

// Async DMA global -> LDS (gfx1250, ASYNCcnt-tracked). Immediate offset applies
// to BOTH the global and LDS addresses (ISA 08_async_tensor §4.4).
// Builtin signature (from hipcc sema): (v4i AS1*, v4i AS3*, imm offset, imm cpol)
#define ASYNC_LD_B128(G, L, OFF)                                        \
  __builtin_amdgcn_global_load_async_to_lds_b128(                       \
      (__attribute__((address_space(1))) v4i*)(G),                      \
      (__attribute__((address_space(3))) v4i*)(L), (OFF), 0)

// order-preserving float -> uint key (works for any sign)
__device__ __forceinline__ unsigned fkey(float f){
  unsigned u = __float_as_uint(f);
  return (u & 0x80000000u) ? ~u : (u | 0x80000000u);
}
__device__ __forceinline__ float keyf(unsigned k){
  unsigned u = (k & 0x80000000u) ? (k & 0x7fffffffu) : ~k;
  return __uint_as_float(u);
}

// ---------------------------------------------------------------------------
// Kernel 1: V[b,h,j,d] = sum_k x[b,j,k] * w[h,k,d], stored transposed/f16 as
// Vt[((b*H+h)*VD + d)*N + j] so kernel 3 B-fragments are contiguous loads.
// WMMA tile: M=16 (j), N=16 (d), K=32; 4 waves = 4 d-tiles of 16.
// ---------------------------------------------------------------------------
__global__ __launch_bounds__(128)
void value_proj_wmma(const float* __restrict__ x, const float* __restrict__ w,
                     _Float16* __restrict__ Vt)
{
  const int tiles = NN/16;
  int idx = blockIdx.x;
  int jt = idx % tiles;
  int h  = (idx / tiles) % HH;
  int b  = idx / (tiles*HH);
  int j0 = jt*16;
  int lane = threadIdx.x & 31;
  int dt   = threadIdx.x >> 5;       // d-tile 0..3
  int half = lane >> 4;
  int lo   = lane & 15;

  const float* xrow = x + (size_t)(b*NN + j0 + lo)*HIDD;   // A row (M = lo)
  v8f acc = {};
  #pragma unroll
  for (int c = 0; c < HIDD/32; ++c){
    v16h a, bf;
    #pragma unroll
    for (int e = 0; e < 16; ++e){
      // A 16-bit layout: lanes<16 hold K=0..7,16..23 ; lanes>=16 hold K=8..15,24..31
      int ka = 32*c + 8*half + (e < 8 ? e : e + 8);
      a[e] = (_Float16)xrow[ka];
      // B 16-bit layout: lane = column d; same K order as A
      int kb = 32*c + 8*half + (e < 8 ? e : e + 8);
      bf[e] = (_Float16)w[((size_t)h*HIDD + kb)*VDD + dt*16 + lo];
    }
    acc = __builtin_amdgcn_wmma_f32_16x16x32_f16(false, a, false, bf,
                                                 (short)0, acc, false, false);
  }
  // D layout: N = lo, M = q + 8*half  ->  Vt[d][j]
  _Float16* dst = Vt + ((size_t)((b*HH + h)*VDD) + dt*16 + lo)*NN;
  #pragma unroll
  for (int q = 0; q < 8; ++q)
    dst[j0 + q + 8*half] = (_Float16)acc[q];
}

// ---------------------------------------------------------------------------
// Kernel 2: per (b,row): exact 30th-percentile threshold via 4-pass radix
// select on 32-bit keys, row min, and per-head softmax denominators
//   S_h = sum_{m<=thresh} exp(-r_h^2*(m - m_min))
// ---------------------------------------------------------------------------
__global__ __launch_bounds__(256)
void row_stats(const float* __restrict__ m, const float* __restrict__ r,
               float* __restrict__ thr, float* __restrict__ mmn,
               float* __restrict__ S)
{
  __shared__ float    s_row[NN];
  __shared__ unsigned s_hist[256];
  __shared__ unsigned s_scan[256];
  __shared__ float    s_redf[HH*256];
  __shared__ unsigned s_prefix, s_minkey, s_mingt, s_cle;
  __shared__ int      s_rank;
  __shared__ float    s_thresh, s_mmin;

  int tid = threadIdx.x;
  int gr  = blockIdx.x;                       // = b*N + i
  // async DMA the whole row into LDS: 64B per thread = 4 x b128
  {
    const float* gsrc = m + (size_t)gr*NN + tid*16;
    float* lp = &s_row[tid*16];
    ASYNC_LD_B128(gsrc, lp, 0);
    ASYNC_LD_B128(gsrc, lp, 16);
    ASYNC_LD_B128(gsrc, lp, 32);
    ASYNC_LD_B128(gsrc, lp, 48);
  }
  if (tid == 0){ s_minkey = 0xffffffffu; s_mingt = 0xffffffffu; s_cle = 0u; }
  __builtin_amdgcn_s_wait_asynccnt(0);
  __syncthreads();

  // 4-pass MSB-first radix select for 0-based rank RANK0
  unsigned prefix = 0; int rank = RANK0;
  for (int pass = 0; pass < 4; ++pass){
    int shift = 24 - 8*pass;
    s_hist[tid] = 0u;
    __syncthreads();
    for (int e = 0; e < 16; ++e){
      unsigned k = fkey(s_row[tid*16 + e]);
      bool match = (pass == 0) || ((k >> (shift + 8)) == (prefix >> (shift + 8)));
      if (match) atomicAdd(&s_hist[(k >> shift) & 255u], 1u);
    }
    __syncthreads();
    unsigned cnt = s_hist[tid];
    s_scan[tid] = cnt;
    __syncthreads();
    for (int off = 1; off < 256; off <<= 1){
      unsigned add = (tid >= off) ? s_scan[tid - off] : 0u;
      __syncthreads();
      s_scan[tid] += add;
      __syncthreads();
    }
    unsigned incl = s_scan[tid], excl = incl - cnt;
    if ((unsigned)rank >= excl && (unsigned)rank < incl){
      s_prefix = prefix | ((unsigned)tid << shift);   // exactly one thread
      s_rank   = rank - (int)excl;
    }
    __syncthreads();
    prefix = s_prefix; rank = s_rank;
    __syncthreads();
  }
  unsigned kth = prefix;   // exact key of sorted[1228]

  // count(<=kth), min(>kth), global min
  unsigned cl = 0;
  for (int e = 0; e < 16; ++e){
    unsigned k = fkey(s_row[tid*16 + e]);
    atomicMin(&s_minkey, k);
    if (k <= kth) cl++; else atomicMin(&s_mingt, k);
  }
  atomicAdd(&s_cle, cl);
  __syncthreads();
  if (tid == 0){
    float v0 = keyf(kth);
    float v1 = (s_cle >= (unsigned)(RANK0 + 2)) ? v0 : keyf(s_mingt);
    s_thresh = 0.5f*(v0 + v1);
    s_mmin   = keyf(s_minkey);
  }
  __syncthreads();
  float th = s_thresh, mmv = s_mmin;

  float r2[HH];
  #pragma unroll
  for (int h = 0; h < HH; ++h){ float rv = r[h]; r2[h] = rv*rv; }
  float acc[HH] = {0.f, 0.f, 0.f, 0.f};
  for (int e = 0; e < 16; ++e){
    float mv = s_row[tid*16 + e];
    if (mv <= th){
      float d = mv - mmv;
      #pragma unroll
      for (int h = 0; h < HH; ++h) acc[h] += __expf(-r2[h]*d);
    }
  }
  #pragma unroll
  for (int h = 0; h < HH; ++h) s_redf[h*256 + tid] = acc[h];
  __syncthreads();
  for (int off = 128; off > 0; off >>= 1){
    if (tid < off){
      #pragma unroll
      for (int h = 0; h < HH; ++h) s_redf[h*256 + tid] += s_redf[h*256 + tid + off];
    }
    __syncthreads();
  }
  if (tid == 0){
    thr[gr] = th; mmn[gr] = mmv;
    int b = gr / NN, i = gr % NN;
    #pragma unroll
    for (int h = 0; h < HH; ++h) S[((size_t)(b*HH + h))*NN + i] = s_redf[h*256];
  }
}

// ---------------------------------------------------------------------------
// Kernel 3: out[b,i,h*VD+d] = gelu( (1/S_h[i]) * sum_j kept*exp(-r2(m-mmin)) * V )
// Block = 16-row tile of one batch; 8 waves = 4 heads x 2 K-halves.
// m chunks DMA'd into double-buffered LDS (async-to-LDS) and reused by all 4
// heads; DMA of chunk k+1 overlaps WMMA/exp compute on chunk k.
// ---------------------------------------------------------------------------
__global__ __launch_bounds__(256)
void att_out_wmma(const float* __restrict__ m, const float* __restrict__ r,
                  const _Float16* __restrict__ Vt,
                  const float* __restrict__ thr, const float* __restrict__ mmn,
                  const float* __restrict__ S, float* __restrict__ out)
{
  __shared__ float lds_m[2][16][68];         // double buffer; 68-stride: no bank conflicts
  __shared__ float lds_red[4][32][32];       // partial accs from kpar=1 waves

  const int tiles = NN/16;
  int it = blockIdx.x % tiles;
  int b  = blockIdx.x / tiles;
  int i0 = it*16;
  int tid  = threadIdx.x;
  int lane = tid & 31;
  int wave = tid >> 5;        // 0..7
  int h    = wave & 3;
  int kpar = wave >> 2;       // 0 or 1: even/odd K-chunks
  int half = lane >> 4;
  int lo   = lane & 15;

  float rv = r[h]; float r2 = rv*rv;
  float th = thr[b*NN + i0 + lo];
  float mmv = mmn[b*NN + i0 + lo];

  int lr = tid >> 4;          // 0..15: row for cooperative DMA
  int lc = tid & 15;          // 0..15: 16B column group
  const float* mrow = m + (size_t)(b*NN + i0 + lr)*NN;
  float* myl[2] = { &lds_m[0][lr][4*lc], &lds_m[1][lr][4*lc] };

  v8f acc[4] = {v8f{}, v8f{}, v8f{}, v8f{}};

  // prologue: DMA first 64-column chunk into buffer 0
  ASYNC_LD_B128(mrow + 4*lc, myl[0], 0);

  for (int k = 0; k < NN/64; ++k){
    int c2 = 2*k;
    int cur = k & 1;
    __builtin_amdgcn_s_wait_asynccnt(0);   // my DMA (into buf[cur]) landed in LDS
    __syncthreads();                       // everyone's landed; prev reads of buf[cur^1] done
    if (k + 1 < NN/64){
      ASYNC_LD_B128(mrow + 32*(c2 + 2) + 4*lc, myl[cur ^ 1], 0);   // overlap with compute
      __builtin_prefetch(mrow + 32*(c2 + 4) + 4*lc, 0, 1);         // global_prefetch_b8
    }

    int cc = c2 + kpar;
    // A fragment: masked exp weights for this head (trans-pipe exp co-executes
    // with WMMA which is tracked as a TRANS op on CDNA5)
    v16h a;
    #pragma unroll
    for (int e = 0; e < 16; ++e){
      int K = 16*half + e;
      float mv = lds_m[cur][lo][32*kpar + K];
      float wgt = (mv <= th) ? __expf(-r2*(mv - mmv)) : 0.0f;
      a[e] = (_Float16)wgt;
    }
    // B fragments: contiguous 16 halves per lane from d-major Vt (L2-resident)
    const _Float16* vb = Vt + ((size_t)((b*HH + h)*VDD) + lo)*NN + 32*cc + 16*half;
    v16h b0 = *(const v16h*)(vb + (size_t)0*16*NN);
    v16h b1 = *(const v16h*)(vb + (size_t)1*16*NN);
    v16h b2 = *(const v16h*)(vb + (size_t)2*16*NN);
    v16h b3 = *(const v16h*)(vb + (size_t)3*16*NN);
    acc[0] = __builtin_amdgcn_wmma_f32_16x16x32_f16(false, a, false, b0, (short)0, acc[0], false, false);
    acc[1] = __builtin_amdgcn_wmma_f32_16x16x32_f16(false, a, false, b1, (short)0, acc[1], false, false);
    acc[2] = __builtin_amdgcn_wmma_f32_16x16x32_f16(false, a, false, b2, (short)0, acc[2], false, false);
    acc[3] = __builtin_amdgcn_wmma_f32_16x16x32_f16(false, a, false, b3, (short)0, acc[3], false, false);
  }

  // fold odd-K partials into even-K waves
  __syncthreads();
  if (kpar == 1){
    #pragma unroll
    for (int dt = 0; dt < 4; ++dt)
      #pragma unroll
      for (int q = 0; q < 8; ++q)
        lds_red[h][lane][dt*8 + q] = acc[dt][q];
  }
  __syncthreads();
  if (kpar == 0){
    float sinv[8];
    #pragma unroll
    for (int q = 0; q < 8; ++q)
      sinv[q] = 1.0f / S[((size_t)(b*HH + h))*NN + i0 + q + 8*half];
    #pragma unroll
    for (int dt = 0; dt < 4; ++dt){
      #pragma unroll
      for (int q = 0; q < 8; ++q){
        int i = q + 8*half;                                  // D: M = q + 8*half
        float v = (acc[dt][q] + lds_red[h][lane][dt*8 + q]) * sinv[q];
        float g = 0.5f * v * (1.0f + erff(v * 0.70710678118654752f)); // exact GELU
        out[((size_t)(b*NN + i0 + i))*HIDD + h*VDD + dt*16 + lo] = g;
      }
    }
  }
}

// ---------------------------------------------------------------------------
extern "C" void kernel_launch(void* const* d_in, const int* in_sizes, int n_in,
                              void* d_out, int out_size, void* d_ws, size_t ws_size,
                              hipStream_t stream)
{
  const float* m = (const float*)d_in[0];   // (B,N,N)
  const float* x = (const float*)d_in[1];   // (B,N,HID)
  const float* r = (const float*)d_in[2];   // (1,H,1,1)
  const float* w = (const float*)d_in[3];   // (H,HID,VD)
  float* out = (float*)d_out;               // (B,N,HID)

  char* ws = (char*)d_ws;
  _Float16* Vt = (_Float16*)ws;                               // B*H*VD*N f16 = 4 MiB
  size_t voff = (size_t)BB*HH*VDD*NN*sizeof(_Float16);
  float* thr = (float*)(ws + voff);                           // B*N
  float* mmn = thr + (size_t)BB*NN;                           // B*N
  float* S   = mmn + (size_t)BB*NN;                           // B*H*N
  (void)in_sizes; (void)n_in; (void)out_size; (void)ws_size;

  value_proj_wmma<<<dim3(BB*HH*(NN/16)), dim3(128), 0, stream>>>(x, w, Vt);
  row_stats      <<<dim3(BB*NN),          dim3(256), 0, stream>>>(m, r, thr, mmn, S);
  att_out_wmma   <<<dim3(BB*(NN/16)),     dim3(256), 0, stream>>>(m, r, Vt, thr, mmn, S, out);
}
